// GNN_77816217469103
// MI455X (gfx1250) — matrix operand order repaired and checked by reading, not compiled, not run on hardware
//
#include <hip/hip_runtime.h>
#include <hip/hip_bf16.h>
#include <math.h>

#define IN_FEATS 128
#define NUM_HEADS 4
#define OUT_FEATS 32
#define HF 128          // NUM_HEADS * OUT_FEATS
#define NEG_SLOPE 0.2f

typedef __attribute__((ext_vector_type(16))) _Float16 v16h;
typedef __attribute__((ext_vector_type(8)))  float    v8f;

// ---------------- convert f32 -> f16 ----------------
__global__ void cvt_f32_f16(const float* __restrict__ in, _Float16* __restrict__ out, int n) {
  int t = blockIdx.x * blockDim.x + threadIdx.x;
  if (t < n) out[t] = (_Float16)in[t];
}

// ---------------- init: out = bias, m = -inf, s = 0 ----------------
__global__ void init_buffers(float* __restrict__ m, float* __restrict__ s,
                             float* __restrict__ out, const float* __restrict__ bias, int N) {
  int t = blockIdx.x * blockDim.x + threadIdx.x;
  if (t < N * HF) out[t] = bias[t & (HF - 1)];
  if (t < N * NUM_HEADS) { m[t] = -INFINITY; s[t] = 0.0f; }
}

// ---------------- WMMA GEMM: C[M,128] = A[M,128] * B^T (B row-major [128,128]) ----------------
union Frag { uint4 u[2]; v16h v; };

__global__ __launch_bounds__(256) void gemm_wmma_f16(
    const _Float16* __restrict__ A, const _Float16* __restrict__ B,
    float* __restrict__ C, int mtiles) {
  int wave = (int)((blockIdx.x * (unsigned)blockDim.x + threadIdx.x) >> 5);
  int lane = threadIdx.x & 31;
  int rowTile = wave >> 3;           // 8 column tiles per row tile (128/16)
  int colTile = wave & 7;
  if (rowTile >= mtiles) return;     // uniform per wave -> EXEC stays all-ones for WMMA
  int grp = lane >> 4;               // lane group 0/1
  int lm  = lane & 15;
  const _Float16* arow = A + (size_t)(rowTile * 16 + lm) * IN_FEATS;
  const _Float16* brow = B + (size_t)(colTile * 16 + lm) * IN_FEATS;  // B[k][n] = W[n][k]
  v8f c = {};
#pragma unroll
  for (int kt = 0; kt < IN_FEATS / 32; ++kt) {
    Frag a, b;
    // A 16x32 f16 layout: lane grp 0: K {0..7, 16..23}; grp 1: K {8..15, 24..31}
    const uint4* pa = (const uint4*)(arow + kt * 32 + grp * 8);
    a.u[0] = pa[0];          // K = grp*8 .. grp*8+7
    a.u[1] = pa[2];          // K = 16+grp*8 .. 16+grp*8+7
    // B 32x16 f16 layout: lane grp 0: K {0..7, 16..23}; grp 1: K {8..15, 24..31}, N = lane%16
    const uint4* pb = (const uint4*)(brow + kt * 32 + grp * 8);
    b.u[0] = pb[0];
    b.u[1] = pb[2];
    c = __builtin_amdgcn_wmma_f32_16x16x32_f16(
        /*neg_a=*/false, a.v, /*neg_b=*/false, b.v,
        /*c_mod=*/(short)0, c, /*reuse_a=*/false, /*reuse_b=*/false);
  }
  // C/D layout: VGPR v -> M = v + 8*grp, N = lane%16
  int row0 = rowTile * 16 + grp * 8;
  int col  = colTile * 16 + lm;
#pragma unroll
  for (int v = 0; v < 8; ++v)
    C[(size_t)(row0 + v) * HF + col] = c[v];
}

// ---------------- el/er = sum_d ft[n,h,d] * attn_{l,r}[h,d] ----------------
__global__ void eler_kernel(const float* __restrict__ ft, const float* __restrict__ al,
                            const float* __restrict__ ar, float* __restrict__ el,
                            float* __restrict__ er, int NH) {
  int t = blockIdx.x * blockDim.x + threadIdx.x;
  if (t >= NH) return;
  int h = t & (NUM_HEADS - 1);
  const float* f = ft + (size_t)t * OUT_FEATS;   // n*128 + h*32 == t*32
  float sl = 0.f, sr = 0.f;
#pragma unroll
  for (int d = 0; d < OUT_FEATS; ++d) {
    float v = f[d];
    sl += v * al[h * OUT_FEATS + d];
    sr += v * ar[h * OUT_FEATS + d];
  }
  el[t] = sl; er[t] = sr;
}

// ---------------- float atomic max via int/uint trick ----------------
__device__ __forceinline__ void atomicMaxFloat(float* addr, float v) {
  if (v >= 0.0f) atomicMax((int*)addr, __float_as_int(v));
  else           atomicMin((unsigned int*)addr, __float_as_uint(v));
}

__device__ __forceinline__ float leaky(float v) { return v > 0.f ? v : NEG_SLOPE * v; }

// ---------------- pass 1: segment max over dst ----------------
__global__ void edge_max(const int* __restrict__ src, const int* __restrict__ dst,
                         const float* __restrict__ el, const float* __restrict__ er,
                         float* __restrict__ m, int EH) {
  int t = blockIdx.x * blockDim.x + threadIdx.x;
  if (t >= EH) return;
  int e = t >> 2, h = t & 3;
  int s = src[e], d = dst[e];
  float v = leaky(el[s * NUM_HEADS + h] + er[d * NUM_HEADS + h]);
  atomicMaxFloat(&m[d * NUM_HEADS + h], v);
}

// ---------------- pass 2: segment sum of exp(e - m[dst]) ----------------
__global__ void edge_sum(const int* __restrict__ src, const int* __restrict__ dst,
                         const float* __restrict__ el, const float* __restrict__ er,
                         const float* __restrict__ m, float* __restrict__ ssum, int EH) {
  int t = blockIdx.x * blockDim.x + threadIdx.x;
  if (t >= EH) return;
  int e = t >> 2, h = t & 3;
  int s = src[e], d = dst[e];
  float v = leaky(el[s * NUM_HEADS + h] + er[d * NUM_HEADS + h]);
  float ex = __expf(v - m[d * NUM_HEADS + h]);
  atomicAdd(&ssum[d * NUM_HEADS + h], ex);
}

// ---------------- pass 3: out[dst] += ft[src] * (a + ft[dst])  (fused msg1+msg2) ----------------
__global__ void edge_agg(const int* __restrict__ src, const int* __restrict__ dst,
                         const float* __restrict__ ft, const float* __restrict__ el,
                         const float* __restrict__ er, const float* __restrict__ m,
                         const float* __restrict__ ssum, float* __restrict__ out,
                         long long total) {
  long long t = (long long)blockIdx.x * blockDim.x + threadIdx.x;
  if (t >= total) return;
  int r = (int)(t & (HF - 1));       // h*32 + d
  int e = (int)(t >> 7);
  int h = r >> 5;
  int s = src[e], d = dst[e];
  float v = leaky(el[s * NUM_HEADS + h] + er[d * NUM_HEADS + h]);
  float a = __expf(v - m[d * NUM_HEADS + h]) / ssum[d * NUM_HEADS + h];
  float fs = ft[(size_t)s * HF + r];
  float fd = ft[(size_t)d * HF + r];
  atomicAdd(&out[(size_t)d * HF + r], fs * (a + fd));
}

// ---------------- launcher ----------------
extern "C" void kernel_launch(void* const* d_in, const int* in_sizes, int n_in,
                              void* d_out, int out_size, void* d_ws, size_t ws_size,
                              hipStream_t stream) {
  const float* feat = (const float*)d_in[0];
  const int*   src  = (const int*)d_in[1];
  const int*   dst  = (const int*)d_in[2];
  const float* W    = (const float*)d_in[3];
  const float* al   = (const float*)d_in[4];
  const float* ar   = (const float*)d_in[5];
  const float* bias = (const float*)d_in[6];
  float* out = (float*)d_out;

  const int N = in_sizes[0] / IN_FEATS;   // 50000
  const int E = in_sizes[1];              // 850000

  // workspace carve-up (256B aligned)
  char* ws = (char*)d_ws;
  size_t o = 0;
  auto take = [&](size_t bytes) { void* p = ws + o; o = (o + bytes + 255) & ~(size_t)255; return p; };
  _Float16* feat_h = (_Float16*)take((size_t)N * IN_FEATS * sizeof(_Float16));
  _Float16* Wh     = (_Float16*)take((size_t)HF * IN_FEATS * sizeof(_Float16));
  float*    ft     = (float*)take((size_t)N * HF * sizeof(float));
  float*    el     = (float*)take((size_t)N * NUM_HEADS * sizeof(float));
  float*    er     = (float*)take((size_t)N * NUM_HEADS * sizeof(float));
  float*    mbuf   = (float*)take((size_t)N * NUM_HEADS * sizeof(float));
  float*    sbuf   = (float*)take((size_t)N * NUM_HEADS * sizeof(float));
  (void)ws_size;

  const int TPB = 256;
  // converts + init (independent)
  cvt_f32_f16<<<(N * IN_FEATS + TPB - 1) / TPB, TPB, 0, stream>>>(feat, feat_h, N * IN_FEATS);
  cvt_f32_f16<<<(HF * IN_FEATS + TPB - 1) / TPB, TPB, 0, stream>>>(W, Wh, HF * IN_FEATS);
  init_buffers<<<(N * HF + TPB - 1) / TPB, TPB, 0, stream>>>(mbuf, sbuf, out, bias, N);

  // WMMA projection: one wave per 16x16 tile; 8 col tiles per row tile -> 1 block per row tile
  int mtiles = N / 16;                    // 50000 divisible by 16
  gemm_wmma_f16<<<mtiles, TPB, 0, stream>>>(feat_h, Wh, ft, mtiles);

  eler_kernel<<<(N * NUM_HEADS + TPB - 1) / TPB, TPB, 0, stream>>>(ft, al, ar, el, er, N * NUM_HEADS);

  int EH = E * NUM_HEADS;
  edge_max<<<(EH + TPB - 1) / TPB, TPB, 0, stream>>>(src, dst, el, er, mbuf, EH);
  edge_sum<<<(EH + TPB - 1) / TPB, TPB, 0, stream>>>(src, dst, el, er, mbuf, sbuf, EH);

  long long total = (long long)E * HF;
  long long blocks = (total + TPB - 1) / TPB;
  edge_agg<<<(unsigned)blocks, TPB, 0, stream>>>(src, dst, ft, el, er, mbuf, sbuf, out, total);
}
